// GCN_GAE_34720515620915
// MI455X (gfx1250) — matrix operand order repaired and checked by reading, not compiled, not run on hardware
//
#include <hip/hip_runtime.h>
#include <hip/hip_bf16.h>

typedef __attribute__((ext_vector_type(16))) _Float16 v16h;
typedef __attribute__((ext_vector_type(8)))  float    v8f;
typedef __attribute__((ext_vector_type(4)))  float    v4f;

#define DIM 256

// ---------------------------------------------------------------------------
// Degree accumulation: one thread per edge, float atomics (exact for counts).
// ---------------------------------------------------------------------------
__global__ void degree_kernel(const int* __restrict__ gsrc,
                              const int* __restrict__ gdst,
                              float* __restrict__ deg_out,
                              float* __restrict__ deg_in,
                              int n_edges) {
    int e = blockIdx.x * blockDim.x + threadIdx.x;
    if (e >= n_edges) return;
    atomicAdd(deg_out + gsrc[e], 1.0f);
    atomicAdd(deg_in  + gdst[e], 1.0f);
}

// In-place deg -> rsqrt(max(deg,1))
__global__ void norm_kernel(float* __restrict__ deg_out,
                            float* __restrict__ deg_in, int n_nodes) {
    int i = blockIdx.x * blockDim.x + threadIdx.x;
    if (i >= n_nodes) return;
    deg_out[i] = rsqrtf(fmaxf(deg_out[i], 1.0f));
    deg_in[i]  = rsqrtf(fmaxf(deg_in[i],  1.0f));
}

// ---------------------------------------------------------------------------
// Sparse aggregation: one wave32 per edge, agg[dst][:] += x[src][:]*norm_s[src]
// Lanes stride the 256-dim row -> coalesced 128B gathers + f32 atomics in L2.
// ---------------------------------------------------------------------------
__global__ __launch_bounds__(256)
void scatter_kernel(const float* __restrict__ xin,
                    const float* __restrict__ norm_s,
                    const int* __restrict__ gsrc,
                    const int* __restrict__ gdst,
                    float* __restrict__ agg, int n_edges) {
    int wid  = (blockIdx.x * blockDim.x + threadIdx.x) >> 5;
    if (wid >= n_edges) return;
    int lane = threadIdx.x & 31;
    int s = gsrc[wid];
    int d = gdst[wid];
    float ns = norm_s[s];
    const float* xs = xin + (size_t)s * DIM;
    float*       ad = agg + (size_t)d * DIM;
#pragma unroll
    for (int j = 0; j < 8; ++j) {
        int idx = lane + 32 * j;
        atomicAdd(ad + idx, xs[idx] * ns);
    }
}

// ---------------------------------------------------------------------------
// Pack a 256x256 f32 row-major W into f16 WMMA B-fragment layout:
//  Wp[((t*8+kb)*32+lane)*16 + i] = W[k][n],
//  n = t*16 + (lane&15), k = kb*32 + (lane>>4)*8 + i + (i>>3)*8
// Each wave then loads its B fragment as two contiguous b128s per lane.
// ---------------------------------------------------------------------------
__global__ void packW_kernel(const float* __restrict__ W,
                             _Float16* __restrict__ Wp) {
    int idx  = blockIdx.x * blockDim.x + threadIdx.x;  // 0..65535
    int i    = idx & 15;
    int lane = (idx >> 4) & 31;
    int kb   = (idx >> 9) & 7;
    int t    = idx >> 12;
    int n = t * 16 + (lane & 15);
    int k = kb * 32 + (lane >> 4) * 8 + i + (i >> 3) * 8;
    Wp[idx] = (_Float16)W[k * DIM + n];
}

// ---------------------------------------------------------------------------
// Fused GEMM: out = maybe_relu( (norm_d[:,None]*agg) @ W + b + 0.1*feat )
// Block = 256 threads = 8 waves; block covers 16 rows x 256 cols.
// Wave w computes two 16x16 output tiles (cols 32w..32w+31).
// A fragment built inline from f32 agg (norm_d folded), f32 accum via WMMA.
// ---------------------------------------------------------------------------
__global__ __launch_bounds__(256)
void gemm_kernel(const float* __restrict__ agg,
                 const float* __restrict__ norm_d,
                 const _Float16* __restrict__ Wp,
                 const float* __restrict__ bias,
                 const float* __restrict__ feat,
                 float* __restrict__ out,
                 int n_nodes, int do_relu) {
    int mtile = blockIdx.x;
    int wave  = threadIdx.x >> 5;
    int lane  = threadIdx.x & 31;
    int half  = lane >> 4;
    int m = mtile * 16 + (lane & 15);
    if (m >= n_nodes) m = n_nodes - 1;          // tail clamp (loads only)
    float nd = norm_d[m];
    const float* arow = agg + (size_t)m * DIM;
    int t0 = wave * 2, t1 = wave * 2 + 1;
    const _Float16* wp0 = Wp + (((size_t)t0 * 8) * 32 + lane) * 16;
    const _Float16* wp1 = Wp + (((size_t)t1 * 8) * 32 + lane) * 16;

    v8f c0 = {}; v8f c1 = {};
#pragma unroll
    for (int kb = 0; kb < 8; ++kb) {
        int k0 = kb * 32 + half * 8;
        v4f x0 = *(const v4f*)(arow + k0);
        v4f x1 = *(const v4f*)(arow + k0 + 4);
        v4f x2 = *(const v4f*)(arow + k0 + 16);
        v4f x3 = *(const v4f*)(arow + k0 + 20);
        v16h a;
#pragma unroll
        for (int i = 0; i < 4; ++i) {
            a[i]      = (_Float16)(x0[i] * nd);
            a[i + 4]  = (_Float16)(x1[i] * nd);
            a[i + 8]  = (_Float16)(x2[i] * nd);
            a[i + 12] = (_Float16)(x3[i] * nd);
        }
        v16h b0 = *(const v16h*)(wp0 + (size_t)kb * 32 * 16);
        v16h b1 = *(const v16h*)(wp1 + (size_t)kb * 32 * 16);
        c0 = __builtin_amdgcn_wmma_f32_16x16x32_f16(false, a, false, b0,
                                                    (short)0, c0, false, false);
        c1 = __builtin_amdgcn_wmma_f32_16x16x32_f16(false, a, false, b1,
                                                    (short)0, c1, false, false);
    }

    // Epilogue: lane holds (row = r + 8*half, col = lane&15) of each tile.
    int col0 = t0 * 16 + (lane & 15);
    int col1 = t1 * 16 + (lane & 15);
    float bia0 = bias[col0], bia1 = bias[col1];
#pragma unroll
    for (int r = 0; r < 8; ++r) {
        int rr = mtile * 16 + r + 8 * half;
        if (rr >= n_nodes) continue;
        float v0 = c0[r] + bia0 + 0.1f * feat[(size_t)rr * DIM + col0];
        float v1 = c1[r] + bia1 + 0.1f * feat[(size_t)rr * DIM + col1];
        if (do_relu) { v0 = fmaxf(v0, 0.0f); v1 = fmaxf(v1, 0.0f); }
        out[(size_t)rr * DIM + col0] = v0;
        out[(size_t)rr * DIM + col1] = v1;
    }
}

// ---------------------------------------------------------------------------
// Edge classifier: wave per query edge.
// logits = [h[s] ; h[t]] @ Wc + bc ; out = sigmoid(logits)
// ---------------------------------------------------------------------------
__global__ __launch_bounds__(256)
void classify_kernel(const float* __restrict__ h,
                     const int* __restrict__ esrc,
                     const int* __restrict__ edst,
                     const float* __restrict__ Wc,   // [512,2] row-major
                     const float* __restrict__ bc,   // [2]
                     float* __restrict__ out, int n_q) {
    int wid = (blockIdx.x * blockDim.x + threadIdx.x) >> 5;
    if (wid >= n_q) return;
    int lane = threadIdx.x & 31;
    int s = esrc[wid], t = edst[wid];
    const float* hs = h + (size_t)s * DIM;
    const float* ht = h + (size_t)t * DIM;
    float a0 = 0.0f, a1 = 0.0f;
#pragma unroll
    for (int j = 0; j < 8; ++j) {
        int idx = lane + 32 * j;
        float v = hs[idx];
        a0 += v * Wc[idx * 2 + 0];
        a1 += v * Wc[idx * 2 + 1];
        float w = ht[idx];
        a0 += w * Wc[(DIM + idx) * 2 + 0];
        a1 += w * Wc[(DIM + idx) * 2 + 1];
    }
#pragma unroll
    for (int off = 16; off > 0; off >>= 1) {
        a0 += __shfl_xor(a0, off, 32);
        a1 += __shfl_xor(a1, off, 32);
    }
    if (lane == 0) {
        float l0 = a0 + bc[0], l1 = a1 + bc[1];
        out[(size_t)wid * 2 + 0] = 1.0f / (1.0f + __expf(-l0));
        out[(size_t)wid * 2 + 1] = 1.0f / (1.0f + __expf(-l1));
    }
}

// ---------------------------------------------------------------------------
extern "C" void kernel_launch(void* const* d_in, const int* in_sizes, int n_in,
                              void* d_out, int out_size, void* d_ws, size_t ws_size,
                              hipStream_t stream) {
    const float* feat = (const float*)d_in[0];
    const float* W1   = (const float*)d_in[1];
    const float* b1   = (const float*)d_in[2];
    const float* W2   = (const float*)d_in[3];
    const float* b2   = (const float*)d_in[4];
    const float* Wc   = (const float*)d_in[5];
    const float* bc   = (const float*)d_in[6];
    const int* g_src  = (const int*)d_in[7];
    const int* g_dst  = (const int*)d_in[8];
    const int* e_src  = (const int*)d_in[9];
    const int* e_dst  = (const int*)d_in[10];

    const int n_nodes = in_sizes[0] / DIM;
    const int n_gedge = in_sizes[7];
    const int n_qedge = in_sizes[9];

    // Workspace layout
    char* ws = (char*)d_ws;
    float*    norm_s = (float*)ws;                               // [N]
    float*    norm_d = norm_s + n_nodes;                         // [N]
    float*    agg    = norm_d + n_nodes;                         // [N*256]
    float*    h1     = agg + (size_t)n_nodes * DIM;              // [N*256]
    _Float16* W1p    = (_Float16*)(h1 + (size_t)n_nodes * DIM);  // [65536]
    _Float16* W2p    = W1p + 65536;                              // [65536]

    float* h_out = (float*)d_out;                     // [N*256]
    float* p_out = h_out + (size_t)n_nodes * DIM;     // [Q*2]

    const int mtiles = (n_nodes + 15) / 16;

    // Pack weights into WMMA B-fragment layout (cheap, every call).
    packW_kernel<<<65536 / 256, 256, 0, stream>>>(W1, W1p);
    packW_kernel<<<65536 / 256, 256, 0, stream>>>(W2, W2p);

    // Degrees -> norms
    hipMemsetAsync(norm_s, 0, (size_t)2 * n_nodes * sizeof(float), stream);
    degree_kernel<<<(n_gedge + 255) / 256, 256, 0, stream>>>(
        g_src, g_dst, norm_s, norm_d, n_gedge);
    norm_kernel<<<(n_nodes + 255) / 256, 256, 0, stream>>>(
        norm_s, norm_d, n_nodes);

    // ---- conv1: agg = scatter(feat*ns); h1 = relu(nd*agg @ W1 + b1 + 0.1f) ----
    hipMemsetAsync(agg, 0, (size_t)n_nodes * DIM * sizeof(float), stream);
    scatter_kernel<<<(n_gedge * 32 + 255) / 256, 256, 0, stream>>>(
        feat, norm_s, g_src, g_dst, agg, n_gedge);
    gemm_kernel<<<mtiles, 256, 0, stream>>>(
        agg, norm_d, W1p, b1, feat, h1, n_nodes, /*relu=*/1);

    // ---- conv2: agg = scatter(h1*ns); h = nd*agg @ W2 + b2 + 0.1f ----
    hipMemsetAsync(agg, 0, (size_t)n_nodes * DIM * sizeof(float), stream);
    scatter_kernel<<<(n_gedge * 32 + 255) / 256, 256, 0, stream>>>(
        h1, norm_s, g_src, g_dst, agg, n_gedge);
    gemm_kernel<<<mtiles, 256, 0, stream>>>(
        agg, norm_d, W2p, b2, feat, h_out, n_nodes, /*relu=*/0);

    // ---- edge gather-concat-classify -> sigmoid ----
    classify_kernel<<<(n_qedge * 32 + 255) / 256, 256, 0, stream>>>(
        h_out, e_src, e_dst, Wc, bc, p_out, n_qedge);
}
